// InterNet_26036091748507
// MI455X (gfx1250) — compile-verified, matrix operand order checked by reading, not run on hardware
//
#include <hip/hip_runtime.h>
#include <hip/hip_bf16.h>

typedef __attribute__((ext_vector_type(16))) __bf16        bf16x16;
typedef __attribute__((ext_vector_type(8)))  float         f32x8;
typedef __attribute__((ext_vector_type(4)))  unsigned int  u32x4;

union FragBF { bf16x16 v; u32x4 q[2]; };

__device__ __forceinline__ f32x8 zero8() {
    f32x8 z = {0.f, 0.f, 0.f, 0.f, 0.f, 0.f, 0.f, 0.f};
    return z;
}

// fp32 -> bf16 round-to-nearest-even
__device__ __forceinline__ unsigned short f2bf(float f) {
    unsigned int u = __float_as_uint(f);
    u += 0x7FFFu + ((u >> 16) & 1u);
    return (unsigned short)(u >> 16);
}

// ---------------------------------------------------------------------------
// Prep: x [img][c][px] fp32  ->  xb [img][px][c] bf16  (pixel-major rows)
// ---------------------------------------------------------------------------
__global__ void k_prep_x(const float* __restrict__ x, unsigned short* __restrict__ xb) {
    int gid = blockIdx.x * blockDim.x + threadIdx.x;
    if (gid >= 32 * 16 * 64 * 64) return;
    int c   = gid & 63;
    int px  = (gid >> 6) & 63;
    int img = gid >> 12;
    xb[gid] = f2bf(x[(img * 64 + c) * 64 + px]);
}

// ---------------------------------------------------------------------------
// Prep: OIHW weights -> per-lane-contiguous WMMA B-fragments, bf16.
// k = p9*cin + c  (p9 = kh*3+kw).  flat = ((t*4+ntile)*32+lane)*16 + e
// lane element e <-> k_local = (e/8)*16 + (lane/16)*8 + (e&7), n = ntile*16+lane%16
// ---------------------------------------------------------------------------
__global__ void k_prep_w(const float* __restrict__ W, unsigned short* __restrict__ wswz, int cin) {
    int gid = blockIdx.x * blockDim.x + threadIdx.x;
    int total = cin * 9 * 64;
    if (gid >= total) return;
    int e     = gid & 15;
    int lane  = (gid >> 4) & 31;
    int ntile = (gid >> 9) & 3;
    int t     = gid >> 11;
    int k = t * 32 + ((e >> 3) << 4) + ((lane >> 4) << 3) + (e & 7);
    int n = ntile * 16 + (lane & 15);
    int p9 = k / cin;
    int c  = k % cin;
    int kh = p9 / 3, kw = p9 % 3;
    wswz[gid] = f2bf(W[((n * cin + c) * 3 + kh) * 3 + kw]);
}

// ---------------------------------------------------------------------------
// Implicit-GEMM 3x3 SAME conv for one wave: 16-pixel M-strip x 64 outputs.
// sInp: LDS image [65 px][CIN ch] bf16 (row 64 is all-zero padding row).
// wswz: swizzled weights (global).  Double-buffered software pipeline:
// fragments of kstep t+1 are in flight while the 4 WMMAs of kstep t issue.
// ---------------------------------------------------------------------------
template <int CIN>
__device__ __forceinline__ void gemm_conv(const unsigned short* sInp,
                                          const unsigned short* __restrict__ wswz,
                                          f32x8 acc[4], int lane, int wv) {
    constexpr int KB = CIN / 32;
    constexpr int T  = KB * 9;
    const int half  = lane >> 4;
    const int pixel = wv * 16 + (lane & 15);      // A-matrix M = lane%16
    const int y  = pixel >> 3, xx = pixel & 7;

    int rowoff[9];                                // per-tap source row offset
#pragma unroll
    for (int p9 = 0; p9 < 9; ++p9) {
        int y2 = y + p9 / 3 - 1, x2 = xx + p9 % 3 - 1;
        bool valid = ((unsigned)y2 < 8u) && ((unsigned)x2 < 8u);
        rowoff[p9] = (valid ? (y2 * 8 + x2) : 64) * CIN;   // row 64 == zeros
    }
    const unsigned short* wl = wswz + (size_t)lane * 16;

    FragBF a[2];
    FragBF bm[2][4];
    auto loadA = [&](int t, int buf) {
        const int p9 = t / KB, kb = t % KB;
        const unsigned short* rp = sInp + rowoff[p9] + kb * 32 + half * 8;
        a[buf].q[0] = *(const u32x4*)(rp);
        a[buf].q[1] = *(const u32x4*)(rp + 16);
    };
    auto loadB = [&](int t, int buf) {
#pragma unroll
        for (int nt = 0; nt < 4; ++nt) {
            const unsigned short* wp = wl + (size_t)(t * 4 + nt) * 512;
            bm[buf][nt].q[0] = *(const u32x4*)(wp);
            bm[buf][nt].q[1] = *(const u32x4*)(wp + 8);
        }
    };

    loadA(0, 0);
    loadB(0, 0);
#pragma unroll
    for (int t = 0; t < T; ++t) {
        const int cur = t & 1, nxt = cur ^ 1;
        if (t + 1 < T) { loadA(t + 1, nxt); loadB(t + 1, nxt); }
#pragma unroll
        for (int nt = 0; nt < 4; ++nt)
            acc[nt] = __builtin_amdgcn_wmma_f32_16x16x32_bf16(
                false, a[cur].v, false, bm[cur][nt].v, (short)0, acc[nt], false, false);
    }
}

// ---------------------------------------------------------------------------
// Fused network: one block per (b, object). 128 threads = 4 waves.
// ---------------------------------------------------------------------------
__global__ void __launch_bounds__(128)
k_fused(const unsigned short* __restrict__ xb, const int* __restrict__ g_idx,
        const unsigned short* __restrict__ wrel, const unsigned short* __restrict__ wself,
        const unsigned short* __restrict__ waff, const unsigned short* __restrict__ wagg,
        const float* __restrict__ brel, const float* __restrict__ bself,
        const float* __restrict__ baff, const float* __restrict__ bagg,
        float* __restrict__ out) {
    __shared__ __align__(16) unsigned short sIn[65 * 128];  // pair / agg input (+zero row)
    __shared__ __align__(16) unsigned short sX [65 * 64];   // x copy, then pred (+zero row)

    const int tid  = threadIdx.x;
    const int lane = tid & 31;
    const int wv   = tid >> 5;
    const int img  = blockIdx.x;      // b*16 + n
    const int b    = img >> 4;
    const int n    = img & 15;
    const int cnl  = lane & 15;       // C/D-matrix N = lane%16
    const int half = lane >> 4;

    // zero the padding rows once
    if (tid < 64) ((unsigned int*)(sIn + 64 * 128))[tid] = 0u;
    if (tid < 32) ((unsigned int*)(sX  + 64 * 64 ))[tid] = 0u;
    __syncthreads();

    float brl[4], bsl[4], bal[4], bgl[4];
#pragma unroll
    for (int nt = 0; nt < 4; ++nt) {
        brl[nt] = brel [nt * 16 + cnl];
        bsl[nt] = bself[nt * 16 + cnl];
        bal[nt] = baff [nt * 16 + cnl];
        bgl[nt] = bagg [nt * 16 + cnl];
    }

    f32x8 accsum[4];
#pragma unroll
    for (int nt = 0; nt < 4; ++nt) accsum[nt] = zero8();

    // ---- relation conv + ReLU + segment-sum over this object's 15 pairs ----
    for (int j = 0; j < 15; ++j) {
        const int gbase = (b * 240 + n * 15 + j) * 3;
        const int g0 = g_idx[gbase + 0];
        const int g1 = g_idx[gbase + 1];
        const int m  = g_idx[gbase + 2];
        if (m == 0) {  // masked pair: conv(0)+bias -> ReLU(bias) everywhere
#pragma unroll
            for (int nt = 0; nt < 4; ++nt) {
                float v = fmaxf(brl[nt], 0.f);
#pragma unroll
                for (int e = 0; e < 8; ++e) accsum[nt][e] += v;
            }
            continue;
        }
        const unsigned int* y1 = (const unsigned int*)(xb + (size_t)(b * 16 + (g0 & 15)) * 4096);
        const unsigned int* y2 = (const unsigned int*)(xb + (size_t)(b * 16 + (g1 & 15)) * 4096);
        unsigned int* s32 = (unsigned int*)sIn;
        __syncthreads();                              // previous GEMM done with sIn
#pragma unroll
        for (int i = 0; i < 32; ++i) {                // 4096 dwords: [px][128ch]
            int word = i * 128 + tid;
            int px = word >> 6;
            int cw = word & 63;                       // dword column (2 channels)
            s32[word] = (cw < 32) ? y1[px * 32 + cw] : y2[px * 32 + (cw - 32)];
        }
        __syncthreads();
        f32x8 acc[4];
#pragma unroll
        for (int nt = 0; nt < 4; ++nt) acc[nt] = zero8();
        gemm_conv<128>(sIn, wrel, acc, lane, wv);
#pragma unroll
        for (int nt = 0; nt < 4; ++nt)
#pragma unroll
            for (int e = 0; e < 8; ++e)
                accsum[nt][e] += fmaxf(acc[nt][e] + brl[nt], 0.f);
    }

    // ---- self conv on original x ----
    {
        const unsigned int* xi = (const unsigned int*)(xb + (size_t)img * 4096);
        unsigned int* sx32 = (unsigned int*)sX;
        __syncthreads();
#pragma unroll
        for (int i = 0; i < 16; ++i) sx32[i * 128 + tid] = xi[i * 128 + tid];
        __syncthreads();
    }
    f32x8 accS[4];
#pragma unroll
    for (int nt = 0; nt < 4; ++nt) accS[nt] = zero8();
    gemm_conv<64>(sX, wself, accS, lane, wv);

    // pred = ReLU(self+b) + rel_sum  -> sX (bf16)
    __syncthreads();
#pragma unroll
    for (int nt = 0; nt < 4; ++nt)
#pragma unroll
        for (int e = 0; e < 8; ++e) {
            float pred = fmaxf(accS[nt][e] + bsl[nt], 0.f) + accsum[nt][e];
            int pixel = wv * 16 + half * 8 + e;       // C/D-matrix M = e + 8*half
            sX[pixel * 64 + nt * 16 + cnl] = f2bf(pred);
        }
    __syncthreads();

    // ---- affinity conv on pred ----
    f32x8 accA[4];
#pragma unroll
    for (int nt = 0; nt < 4; ++nt) accA[nt] = zero8();
    gemm_conv<64>(sX, waff, accA, lane, wv);

    // build agg input in sIn: ch 0..63 = ReLU(aff+b), ch 64..127 = original x
    __syncthreads();
    {
        const unsigned int* xi = (const unsigned int*)(xb + (size_t)img * 4096);
        unsigned int* s32 = (unsigned int*)sIn;
#pragma unroll
        for (int i = 0; i < 16; ++i) {
            int idx = i * 128 + tid;                  // 2048 dwords of x
            int px = idx >> 5;
            int cw = idx & 31;
            s32[px * 64 + 32 + cw] = xi[px * 32 + cw];
        }
    }
#pragma unroll
    for (int nt = 0; nt < 4; ++nt)
#pragma unroll
        for (int e = 0; e < 8; ++e) {
            float a = fmaxf(accA[nt][e] + bal[nt], 0.f);
            int pixel = wv * 16 + half * 8 + e;
            sIn[pixel * 128 + nt * 16 + cnl] = f2bf(a);
        }
    __syncthreads();

    // ---- aggregation conv, write fp32 NCHW output ----
    f32x8 accG[4];
#pragma unroll
    for (int nt = 0; nt < 4; ++nt) accG[nt] = zero8();
    gemm_conv<128>(sIn, wagg, accG, lane, wv);

    float* outp = out + (size_t)img * 4096;
#pragma unroll
    for (int nt = 0; nt < 4; ++nt)
#pragma unroll
        for (int e = 0; e < 8; ++e) {
            int pixel = wv * 16 + half * 8 + e;
            outp[(nt * 16 + cnl) * 64 + pixel] = fmaxf(accG[nt][e] + bgl[nt], 0.f);
        }
}

// ---------------------------------------------------------------------------
extern "C" void kernel_launch(void* const* d_in, const int* in_sizes, int n_in,
                              void* d_out, int out_size, void* d_ws, size_t ws_size,
                              hipStream_t stream) {
    (void)in_sizes; (void)n_in; (void)out_size; (void)ws_size;
    const float* x      = (const float*)d_in[0];
    const int*   g_idx  = (const int*)  d_in[1];
    const float* W_rel  = (const float*)d_in[2];
    const float* b_rel  = (const float*)d_in[3];
    const float* W_self = (const float*)d_in[4];
    const float* b_self = (const float*)d_in[5];
    const float* W_aff  = (const float*)d_in[6];
    const float* b_aff  = (const float*)d_in[7];
    const float* W_agg  = (const float*)d_in[8];
    const float* b_agg  = (const float*)d_in[9];
    float* out = (float*)d_out;

    unsigned short* xb    = (unsigned short*)d_ws;      // 2,097,152 bf16
    unsigned short* wrel  = xb    + 2097152;            // 73,728
    unsigned short* wself = wrel  + 73728;              // 36,864
    unsigned short* waff  = wself + 36864;              // 36,864
    unsigned short* wagg  = waff  + 36864;              // 73,728  (total ~4.6 MB)

    k_prep_x<<<(2097152 + 255) / 256, 256, 0, stream>>>(x, xb);
    k_prep_w<<<(73728 + 255) / 256, 256, 0, stream>>>(W_rel,  wrel,  128);
    k_prep_w<<<(36864 + 255) / 256, 256, 0, stream>>>(W_self, wself, 64);
    k_prep_w<<<(36864 + 255) / 256, 256, 0, stream>>>(W_aff,  waff,  64);
    k_prep_w<<<(73728 + 255) / 256, 256, 0, stream>>>(W_agg,  wagg,  128);
    k_fused<<<512, 128, 0, stream>>>(xb, g_idx, wrel, wself, waff, wagg,
                                     b_rel, b_self, b_aff, b_agg, out);
}
